// MEMO_22514218566221
// MI455X (gfx1250) — matrix operand order repaired and checked by reading, not compiled, not run on hardware
//
#include <hip/hip_runtime.h>
#include <hip/hip_bf16.h>

// ---------------------------------------------------------------------------
// Problem constants (from reference)
// ---------------------------------------------------------------------------
#define Bn    65536
#define OBSn  512
#define OUTn  64
#define Hn    1024
#define Kn    10
#define TESTn 100
#define LOG2PI 1.8378770664093453f

typedef __bf16        v16bf  __attribute__((ext_vector_type(16)));
typedef float         v8f    __attribute__((ext_vector_type(8)));
typedef unsigned int  u32x4  __attribute__((ext_vector_type(4)));
typedef unsigned int  u32x8  __attribute__((ext_vector_type(8)));

union FragBF { v16bf v; unsigned int u[8]; };

// Tensor Data Mover staging for the A tile (gfx1250 device pass only).
#if defined(__gfx1250__)
#define USE_TDM_A 1
#else
#define USE_TDM_A 0
#endif

__device__ __forceinline__ unsigned short f2bf(float f) {
    unsigned int u = __builtin_bit_cast(unsigned int, f);
    unsigned int r = u + 0x7FFFu + ((u >> 16) & 1u);
    return (unsigned short)(r >> 16);
}
__device__ __forceinline__ float bf2f(unsigned short h) {
    unsigned int u = ((unsigned int)h) << 16;
    return __builtin_bit_cast(float, u);
}

#if USE_TDM_A
__device__ __forceinline__ void wait_tensorcnt0() {
#if __has_builtin(__builtin_amdgcn_s_wait_tensorcnt)
    __builtin_amdgcn_s_wait_tensorcnt(0);
#else
    asm volatile("s_wait_tensorcnt 0x0" ::: "memory");
#endif
}

// One TDM descriptor: copy a 128-row x 32-element (bf16) 2D tile, row stride
// lda elements, into LDS with 16B padding after each 64B row (-> 80B stride).
// ISA 8.3/8.4: D# group0 = {type=2, global_addr[56:0], lds_addr, count=1},
// group1 = {strides, tile dims, tensor dims, pad ctl, data_size=2B, mask=0}.
__device__ __forceinline__ void tdm_load_a_tile(const unsigned short* gbase,
                                                unsigned int ldsOff, int lda) {
    unsigned long long ga = (unsigned long long)(size_t)gbase;
    u32x4 g0;
    g0[0] = 1u;                                                   // count=1 (valid user D#)
    g0[1] = ldsOff;                                               // lds_addr (bytes)
    g0[2] = (unsigned int)ga;                                     // global_addr[31:0]
    g0[3] = (unsigned int)((ga >> 32) & 0x01FFFFFFu) | (2u << 30);// global_addr[56:32], type=2
    u32x8 g1;
    g1[0] = (1u << 16)      // data_size = 1 -> 2 bytes
          | (1u << 20)      // pad_enable
          | (3u << 22)      // pad_interval: pad after 64B of data
          | (3u << 25);     // pad_amount: 4 DWORDs = 16B
    g1[1] = ((unsigned int)lda & 0xFFFFu) << 16;                  // tensor_dim0[15:0]
    g1[2] = ((unsigned int)lda >> 16)                             // tensor_dim0[31:16]
          | (((unsigned int)Bn & 0xFFFFu) << 16);                 // tensor_dim1[15:0]
    g1[3] = ((unsigned int)Bn >> 16) | (32u << 16);               // tensor_dim1[31:16], tile_dim0=32
    g1[4] = 128u;                                                 // tile_dim1=128, tile_dim2=0
    g1[5] = (unsigned int)lda;                                    // tensor_dim0_stride[31:0]
    g1[6] = 0u;                                                   // stride[47:32]=0, dim1_stride lo=0
    g1[7] = 0u;                                                   // dim1_stride hi=0
    // 2D tensor -> 2-group form (VADDR2/3 = NULL)
    asm volatile("tensor_load_to_lds %0, %1" :: "s"(g0), "s"(g1) : "memory");
}
#endif

// ---------------------------------------------------------------------------
// fp32 -> bf16 convert
// ---------------------------------------------------------------------------
__global__ void k_cvt_bf16(const float* __restrict__ s, unsigned short* __restrict__ d, int n) {
    int i = blockIdx.x * blockDim.x + threadIdx.x;
    if (i < n) d[i] = f2bf(s[i]);
}

// ---------------------------------------------------------------------------
// WMMA GEMM: out = act(A[M x K] @ W[K x N] + bias)
//   A, W bf16 (ushort-packed).  Block tile 128x64, wave tile 32x32.
//   A tile staged global->LDS by the Tensor Data Mover (one D# per K-tile,
//   double-buffered, wave 0 issues + waits TENSORcnt, barrier publishes).
//   W tile staged transposed ([n][k], K contiguous) via registers.
//   Row stride 40 halves = 80B: bank-conflict-free over 16 rows, b128-aligned,
//   and expressible as TDM pad (64B data + 16B pad).
//   Callers guarantee A rows are readable (and zero) up to roundup32(K) <= lda.
//   ACT: 0=none 1=tanh 2=relu 3=tanh(relu)
// ---------------------------------------------------------------------------
#define LDS_STRIDE 40

__device__ __forceinline__ void load_a_regs(const unsigned short* __restrict__ Am,
                                            int lda, int Kdim, int rowBase, int k0,
                                            int t, unsigned int (&r)[8]) {
#pragma unroll
    for (int it = 0; it < 8; ++it) {
        int d  = t + it * 256;          // 0..2047 over 128x16 dwords
        int rr = d >> 4;                // row 0..127
        int cp = (d & 15) << 1;         // col pair 0..30
        int gc = k0 + cp;
        const unsigned short* p = Am + (size_t)(rowBase + rr) * lda + gc;
        unsigned int v = 0u;
        if (gc + 1 < Kdim)      v = *(const unsigned int*)p;
        else if (gc < Kdim)     v = (unsigned int)(*p);
        r[it] = v;
    }
}
__device__ __forceinline__ void store_a_lds(unsigned short* __restrict__ As,
                                            int t, const unsigned int (&r)[8]) {
#pragma unroll
    for (int it = 0; it < 8; ++it) {
        int d  = t + it * 256;
        int rr = d >> 4;
        int cp = (d & 15) << 1;
        *(unsigned int*)(As + rr * LDS_STRIDE + cp) = r[it];
    }
}
__device__ __forceinline__ void load_b_regs(const unsigned short* __restrict__ Wm,
                                            int ldw, int Kdim, int N, int colBase, int k0,
                                            int t, unsigned int (&r)[4]) {
#pragma unroll
    for (int it = 0; it < 4; ++it) {
        int d  = t + it * 256;          // 0..1023 over 64x16 dwords
        int n  = d >> 4;                // col 0..63
        int kp = (d & 15) << 1;         // k pair 0..30
        int gn = colBase + n;
        unsigned int v = 0u;
        if (gn < N) {
            unsigned int lo = 0u, hi = 0u;
            if (k0 + kp     < Kdim) lo = Wm[(size_t)(k0 + kp)     * ldw + gn];
            if (k0 + kp + 1 < Kdim) hi = Wm[(size_t)(k0 + kp + 1) * ldw + gn];
            v = lo | (hi << 16);
        }
        r[it] = v;
    }
}
__device__ __forceinline__ void store_b_lds(unsigned short* __restrict__ Bs,
                                            int t, const unsigned int (&r)[4]) {
#pragma unroll
    for (int it = 0; it < 4; ++it) {
        int d  = t + it * 256;
        int n  = d >> 4;
        int kp = (d & 15) << 1;
        *(unsigned int*)(Bs + n * LDS_STRIDE + kp) = r[it];
    }
}

template <int ACT>
__global__ void __launch_bounds__(256)
k_gemm(const unsigned short* __restrict__ Am, const unsigned short* __restrict__ Wm,
       const float* __restrict__ bias,
       unsigned short* __restrict__ outb, float* __restrict__ outf,
       int N, int Kdim, int lda, int ldw, int ldo) {
    __shared__ unsigned short As[2][128 * LDS_STRIDE];
    __shared__ unsigned short Bs[2][64 * LDS_STRIDE];

    const int t       = threadIdx.x;
    const int rowBase = blockIdx.y * 128;
    const int colBase = blockIdx.x * 64;
    const int lane    = t & 31;
    const int wave    = t >> 5;
    const int wm      = (wave & 3) * 32;   // wave row within block tile
    const int wn      = (wave >> 2) * 32;  // wave col within block tile
    const int half    = lane >> 4;
    const int m16     = lane & 15;

    v8f c00 = {}, c01 = {}, c10 = {}, c11 = {};

    const int kTiles = (Kdim + 31) >> 5;

#if USE_TDM_A
    const unsigned int ldsA0 = (unsigned int)(size_t)&As[0][0];
    const unsigned int ldsA1 = (unsigned int)(size_t)&As[1][0];
    if (wave == 0)
        tdm_load_a_tile(Am + (size_t)rowBase * lda, ldsA0, lda);
    unsigned int rb[4];
#else
    unsigned int ra[8];
    unsigned int rb[4];
    load_a_regs(Am, lda, Kdim, rowBase, 0, t, ra);
    store_a_lds(As[0], t, ra);
#endif
    load_b_regs(Wm, ldw, Kdim, N, colBase, 0, t, rb);
    store_b_lds(Bs[0], t, rb);
#if USE_TDM_A
    if (wave == 0) wait_tensorcnt0();
#endif
    __syncthreads();

    for (int kt = 0; kt < kTiles; ++kt) {
        const int cur = kt & 1;
        const int nxt = cur ^ 1;
        if (kt + 1 < kTiles) {
            const int k1 = (kt + 1) << 5;
#if USE_TDM_A
            if (wave == 0)
                tdm_load_a_tile(Am + (size_t)rowBase * lda + k1,
                                nxt ? ldsA1 : ldsA0, lda);
#else
            load_a_regs(Am, lda, Kdim, rowBase, k1, t, ra);
#endif
            load_b_regs(Wm, ldw, Kdim, N, colBase, k1, t, rb);
        }

        FragBF a0, a1, b0, b1;
        const unsigned short* Ap = As[cur];
        const unsigned short* Bp = Bs[cur];
#pragma unroll
        for (int r = 0; r < 8; ++r) {
            // ISA 7.12.2: bf16 A 16x32 — lanes 0-15 hold K 0-7 & 16-23, lanes 16-31 hold K 8-15 & 24-31
            int kb  = (r < 4) ? (half * 8 + 2 * r) : (16 + half * 8 + 2 * (r - 4));
            a0.u[r] = *(const unsigned int*)(Ap + (wm + m16) * LDS_STRIDE + kb);
            a1.u[r] = *(const unsigned int*)(Ap + (wm + 16 + m16) * LDS_STRIDE + kb);
            // B 32x16 — same K order as A: lanes 0-15 hold K 0-7 & 16-23, lanes 16-31 hold K 8-15 & 24-31 (col = lane&15)
            int kbb = (r < 4) ? (half * 8 + 2 * r) : (16 + half * 8 + 2 * (r - 4));
            b0.u[r] = *(const unsigned int*)(Bp + (wn + m16) * LDS_STRIDE + kbb);
            b1.u[r] = *(const unsigned int*)(Bp + (wn + 16 + m16) * LDS_STRIDE + kbb);
        }
        c00 = __builtin_amdgcn_wmma_f32_16x16x32_bf16(false, a0.v, false, b0.v, (short)0, c00, false, false);
        c01 = __builtin_amdgcn_wmma_f32_16x16x32_bf16(false, a0.v, false, b1.v, (short)0, c01, false, false);
        c10 = __builtin_amdgcn_wmma_f32_16x16x32_bf16(false, a1.v, false, b0.v, (short)0, c10, false, false);
        c11 = __builtin_amdgcn_wmma_f32_16x16x32_bf16(false, a1.v, false, b1.v, (short)0, c11, false, false);

        if (kt + 1 < kTiles) {
#if !USE_TDM_A
            store_a_lds(As[nxt], t, ra);
#endif
            store_b_lds(Bs[nxt], t, rb);
        }
#if USE_TDM_A
        if (wave == 0) wait_tensorcnt0();
#endif
        __syncthreads();
    }

    // Epilogue: ISA layout — C/D 16x16 f32: lane l, VGPR r -> M = r + (l>>4)*8, N = l&15
    v8f cs[2][2] = {{c00, c01}, {c10, c11}};
#pragma unroll
    for (int tm = 0; tm < 2; ++tm) {
#pragma unroll
        for (int tn = 0; tn < 2; ++tn) {
            int gn = colBase + wn + tn * 16 + m16;
            if (gn < N) {
                float bv = bias ? bias[gn] : 0.0f;
#pragma unroll
                for (int r = 0; r < 8; ++r) {
                    int   gm = rowBase + wm + tm * 16 + half * 8 + r;
                    float v  = cs[tm][tn][r] + bv;
                    if (ACT == 1) v = tanhf(v);
                    else if (ACT == 2) v = fmaxf(v, 0.0f);
                    else if (ACT == 3) v = tanhf(fmaxf(v, 0.0f));
                    outb[(size_t)gm * ldo + gn] = f2bf(v);
                    if (outf) outf[(size_t)gm * ldo + gn] = v;
                }
            }
        }
    }
}

// ---------------------------------------------------------------------------
// codebook argmin per row:  argmin_k ||c_k||^2 - 2 e.c_k
// ---------------------------------------------------------------------------
__global__ void k_argmin(const unsigned short* __restrict__ enc,
                         const float* __restrict__ cb, int* __restrict__ prop) {
    int row = blockIdx.x * blockDim.x + threadIdx.x;
    if (row >= Bn) return;
    const unsigned short* e = enc + (size_t)row * OBSn;
    float best = 3.4e38f;
    int   bi   = 0;
    for (int k = 0; k < Kn; ++k) {
        const float* c = cb + k * OBSn;
        float s1 = 0.0f, s2 = 0.0f;
        for (int j = 0; j < OBSn; ++j) {
            float cv = c[j];
            float ev = bf2f(e[j]);
            s1 += cv * cv;
            s2 += ev * cv;
        }
        float sc = s1 - 2.0f * s2;
        if (sc < best) { best = sc; bi = k; }
    }
    prop[row] = bi;
}

// gather quantized rows (bf16) + accumulate vq MSE numerator into acc[1]
__global__ void k_quant(const unsigned short* __restrict__ enc,
                        const float* __restrict__ cb, const int* __restrict__ prop,
                        unsigned short* __restrict__ q, float* __restrict__ acc) {
    __shared__ float red[256];
    size_t i   = (size_t)blockIdx.x * blockDim.x + threadIdx.x;
    int    row = (int)(i >> 9);
    int    j   = (int)(i & 511);
    float  c   = cb[prop[row] * OBSn + j];
    q[i]       = f2bf(c);
    float d    = bf2f(enc[i]) - c;
    red[threadIdx.x] = d * d;
    __syncthreads();
    for (int s = 128; s > 0; s >>= 1) {
        if (threadIdx.x < s) red[threadIdx.x] += red[threadIdx.x + s];
        __syncthreads();
    }
    if (threadIdx.x == 0) atomicAdd(&acc[1], red[0]);
}

// recon MSE numerator into acc[0]
__global__ void k_mse_recon(const float* __restrict__ dX,
                            const unsigned short* __restrict__ recon,
                            float* __restrict__ acc) {
    __shared__ float red[256];
    size_t i = (size_t)blockIdx.x * blockDim.x + threadIdx.x;
    float  d = dX[i] - bf2f(recon[i]);
    red[threadIdx.x] = d * d;
    __syncthreads();
    for (int s = 128; s > 0; s >>= 1) {
        if (threadIdx.x < s) red[threadIdx.x] += red[threadIdx.x + s];
        __syncthreads();
    }
    if (threadIdx.x == 0) atomicAdd(&acc[0], red[0]);
}

// actor input: s[row, 0..511]=bf16(X), s[row,512]=bf16(prop), pad to 544 with 0
__global__ void k_build_s(const float* __restrict__ X, const int* __restrict__ prop,
                          unsigned short* __restrict__ s) {
    size_t i = (size_t)blockIdx.x * blockDim.x + threadIdx.x;
    if (i >= (size_t)Bn * 544) return;
    int row = (int)(i / 544);
    int j   = (int)(i % 544);
    float v = (j < OBSn) ? X[(size_t)row * OBSn + j]
            : (j == OBSn ? (float)prop[row] : 0.0f);
    s[i] = f2bf(v);
}

// per-row Normal log-prob -> recon_loss
__global__ void k_logprob(const float* __restrict__ Aact, const float* __restrict__ mu,
                          const float* __restrict__ log_std, float* __restrict__ rl) {
    int row = blockIdx.x * blockDim.x + threadIdx.x;
    if (row >= Bn) return;
    float acc = 0.0f;
    for (int j = 0; j < OUTn; ++j) {
        float ls = log_std[j];
        float z  = (Aact[(size_t)row * OUTn + j] - mu[(size_t)row * OUTn + j]) * expf(-ls);
        acc += -0.5f * z * z - ls - 0.5f * LOG2PI;
    }
    rl[row] = -acc;
}

__global__ void k_copyX(const float* __restrict__ X, float* __restrict__ out) {
    size_t i = (size_t)blockIdx.x * blockDim.x + threadIdx.x;
    out[i] = X[i];
}

__global__ void k_init_acc(float* acc) {
    if (threadIdx.x < 4) acc[threadIdx.x] = 0.0f;
}

__global__ void k_finalize(const float* __restrict__ rl, const int* __restrict__ prop,
                           const float* __restrict__ acc, float* __restrict__ out) {
    int   i        = blockIdx.x * blockDim.x + threadIdx.x;
    float inv      = 1.0f / ((float)Bn * (float)OBSn);
    // forward value: commitment_loss == vq_loss
    float vq_total = (acc[0] + 2.0f * acc[1]) * inv;
    if (i < Bn) {
        out[i]                              = rl[i] * vq_total;   // loss
        out[Bn + i]                         = rl[i];              // recon_loss
        out[2 * Bn + (size_t)Bn * OBSn + i] = (float)prop[i];     // proposal
    }
    if (i == 0) out[2 * Bn + (size_t)Bn * OBSn + Bn] = vq_total;  // vq_total_loss
}

// ---------------------------------------------------------------------------
// launch
// ---------------------------------------------------------------------------
extern "C" void kernel_launch(void* const* d_in, const int* in_sizes, int n_in,
                              void* d_out, int out_size, void* d_ws, size_t ws_size,
                              hipStream_t stream) {
    const float* X        = (const float*)d_in[0];
    const float* dX       = (const float*)d_in[1];
    const float* Aact     = (const float*)d_in[2];
    const float* enc_w1   = (const float*)d_in[4];
    const float* enc_b1   = (const float*)d_in[5];
    const float* enc_w2   = (const float*)d_in[6];
    const float* enc_b2   = (const float*)d_in[7];
    const float* prenet_w = (const float*)d_in[8];
    const float* prenet_b = (const float*)d_in[9];
    const float* codebook = (const float*)d_in[10];
    const float* postnet_w= (const float*)d_in[11];
    const float* postnet_b= (const float*)d_in[12];
    const float* dec_w1   = (const float*)d_in[13];
    const float* dec_b1   = (const float*)d_in[14];
    const float* dec_w2   = (const float*)d_in[15];
    const float* dec_b2   = (const float*)d_in[16];
    const float* a_w1     = (const float*)d_in[17];
    const float* a_b1     = (const float*)d_in[18];
    const float* a_w2     = (const float*)d_in[19];
    const float* a_b2     = (const float*)d_in[20];
    const float* a_w3     = (const float*)d_in[21];
    const float* a_b3     = (const float*)d_in[22];
    const float* a_w4     = (const float*)d_in[23];
    const float* a_b4     = (const float*)d_in[24];
    const float* log_std  = (const float*)d_in[25];

    char*  ws  = (char*)d_ws;
    size_t off = 0;
    auto alloc = [&](size_t bytes) -> char* {
        char* p = ws + off;
        off += (bytes + 255) & ~(size_t)255;
        return p;
    };

    float*          acc   = (float*)alloc(4 * sizeof(float));
    unsigned short* wbE1  = (unsigned short*)alloc((size_t)512 * 512 * 2);
    unsigned short* wbE2  = (unsigned short*)alloc((size_t)512 * 1024 * 2);
    unsigned short* wbPre = (unsigned short*)alloc((size_t)1024 * 512 * 2);
    unsigned short* wbPost= (unsigned short*)alloc((size_t)512 * 1024 * 2);
    unsigned short* wbD1  = (unsigned short*)alloc((size_t)1024 * 100 * 2);
    unsigned short* wbD2  = (unsigned short*)alloc((size_t)100 * 512 * 2);
    unsigned short* wbA1  = (unsigned short*)alloc((size_t)513 * 512 * 2);
    unsigned short* wbA2  = (unsigned short*)alloc((size_t)512 * 512 * 2);
    unsigned short* wbA3  = (unsigned short*)alloc((size_t)512 * 512 * 2);
    unsigned short* wbA4  = (unsigned short*)alloc((size_t)512 * 64 * 2);

    unsigned short* dXb   = (unsigned short*)alloc((size_t)Bn * 512 * 2);   // later: m1
    unsigned short* hB    = (unsigned short*)alloc((size_t)Bn * 512 * 2);   // later: d1 (B x 128)
    unsigned short* midC  = (unsigned short*)alloc((size_t)Bn * 1024 * 2);  // later: post
    unsigned short* encD  = (unsigned short*)alloc((size_t)Bn * 512 * 2);   // later: m2
    unsigned short* qE    = (unsigned short*)alloc((size_t)Bn * 512 * 2);   // later: m3
    unsigned short* reconF= (unsigned short*)alloc((size_t)Bn * 512 * 2);
    unsigned short* sG    = (unsigned short*)alloc((size_t)Bn * 544 * 2);   // later: dummy mu_bf
    float*          muH   = (float*)alloc((size_t)Bn * 64 * 4);
    int*            prop  = (int*)alloc((size_t)Bn * 4);
    float*          rl    = (float*)alloc((size_t)Bn * 4);
    (void)ws_size; (void)in_sizes; (void)n_in; (void)out_size;

    float* out = (float*)d_out;

    auto cvt = [&](const float* s, unsigned short* d, int n) {
        k_cvt_bf16<<<(n + 255) / 256, 256, 0, stream>>>(s, d, n);
    };
    cvt(enc_w1, wbE1, 512 * 512);
    cvt(enc_w2, wbE2, 512 * 1024);
    cvt(prenet_w, wbPre, 1024 * 512);
    cvt(postnet_w, wbPost, 512 * 1024);
    cvt(dec_w1, wbD1, 1024 * 100);
    cvt(dec_w2, wbD2, 100 * 512);
    cvt(a_w1, wbA1, 513 * 512);
    cvt(a_w2, wbA2, 512 * 512);
    cvt(a_w3, wbA3, 512 * 512);
    cvt(a_w4, wbA4, 512 * 64);
    k_cvt_bf16<<<(Bn * 512 + 255) / 256, 256, 0, stream>>>(dX, dXb, Bn * 512);
    k_init_acc<<<1, 32, 0, stream>>>(acc);

    const dim3 blk(256);
    auto gdim = [](int N) { return dim3((N + 63) / 64, Bn / 128); };

    // VQ encoder + prenet
    k_gemm<1><<<gdim(512), blk, 0, stream>>>(dXb, wbE1, enc_b1, hB, nullptr, 512, 512, 512, 512, 512);
    k_gemm<0><<<gdim(1024), blk, 0, stream>>>(hB, wbE2, enc_b2, midC, nullptr, 1024, 512, 512, 1024, 1024);
    k_gemm<0><<<gdim(512), blk, 0, stream>>>(midC, wbPre, prenet_b, encD, nullptr, 512, 1024, 1024, 512, 512);

    // quantize
    k_argmin<<<Bn / 256, 256, 0, stream>>>(encD, codebook, prop);
    k_quant<<<(Bn * 512) / 256, 256, 0, stream>>>(encD, codebook, prop, qE, acc);

    // postnet + decoder
    unsigned short* post = midC;
    unsigned short* d1   = hB;     // ld = 128, K = 100; zero-filled so TDM edge tiles read 0
    k_gemm<0><<<gdim(1024), blk, 0, stream>>>(qE, wbPost, postnet_b, post, nullptr, 1024, 512, 512, 1024, 1024);
    (void)hipMemsetAsync(d1, 0, (size_t)Bn * 128 * 2, stream);
    k_gemm<1><<<gdim(100), blk, 0, stream>>>(post, wbD1, dec_b1, d1, nullptr, 100, 1024, 1024, 100, 128);
    k_gemm<1><<<gdim(512), blk, 0, stream>>>(d1, wbD2, dec_b2, reconF, nullptr, 512, 100, 128, 512, 512);
    k_mse_recon<<<(Bn * 512) / 256, 256, 0, stream>>>(dX, reconF, acc);

    // actor
    k_build_s<<<((size_t)Bn * 544 + 255) / 256, 256, 0, stream>>>(X, prop, sG);
    unsigned short* m1 = dXb;
    unsigned short* m2 = encD;
    unsigned short* m3 = qE;
    k_gemm<2><<<gdim(512), blk, 0, stream>>>(sG, wbA1, a_b1, m1, nullptr, 512, 513, 544, 512, 512);
    k_gemm<2><<<gdim(512), blk, 0, stream>>>(m1, wbA2, a_b2, m2, nullptr, 512, 512, 512, 512, 512);
    k_gemm<3><<<gdim(512), blk, 0, stream>>>(m2, wbA3, a_b3, m3, nullptr, 512, 512, 512, 512, 512);
    k_gemm<0><<<gdim(64), blk, 0, stream>>>(m3, wbA4, a_b4, sG /*dummy bf out*/, muH, 64, 512, 512, 64, 64);

    // losses + outputs
    k_logprob<<<Bn / 256, 256, 0, stream>>>(Aact, muH, log_std, rl);
    k_copyX<<<(Bn * 512) / 256, 256, 0, stream>>>(X, out + 2 * Bn);
    k_finalize<<<Bn / 256, 256, 0, stream>>>(rl, prop, acc, out);
}
